// FusedBlock_18597208392420
// MI455X (gfx1250) — matrix operand order repaired and checked by reading, not compiled, not run on hardware
//
#include <hip/hip_runtime.h>

// ---------------- problem constants ----------------
#define S_    2048
#define D_    2048
#define H_    16
#define HD_   128
#define DV_   128
#define BS_   512
#define NB_   4        // S_/BS_
#define KSEL  2
#define R_    64
#define DFF_  8192
#define CT_   1600     // BS + KSEL*BS + R
#define QT_   32       // q rows per attention workgroup
#define EPS_  1e-6f
#define TAU_  0.2f
#define NEG_  -1e9f
#define SCALE_ 0.08838834764831845f  // 1/sqrt(128)

typedef __bf16 bf16_t;
typedef bf16_t bf16x16 __attribute__((ext_vector_type(16)));
typedef float  f32x8   __attribute__((ext_vector_type(8)));

__device__ __forceinline__ f32x8 f32x8_zero() {
  f32x8 z;
#pragma unroll
  for (int i = 0; i < 8; ++i) z[i] = 0.f;
  return z;
}

__device__ __forceinline__ f32x8 wmma_bf16(bf16x16 a, bf16x16 b, f32x8 c) {
  return __builtin_amdgcn_wmma_f32_16x16x32_bf16(false, a, false, b, (short)0, c,
                                                 false, false);
}

// A-fragment (16x32 bf16): per lane one contiguous 16-elem run:
// K = hl*16 + e   (matches contiguous B fragments)
__device__ __forceinline__ bf16x16 frag_a(const bf16_t* rowptr, int hl) {
  bf16x16 a;
#pragma unroll
  for (int e = 0; e < 16; ++e) {
    int kk = (hl << 4) + e;
    a[e] = rowptr[kk];
  }
  return a;
}
// B-fragment from K-contiguous storage: 16 contiguous elems at + hl*16
__device__ __forceinline__ bf16x16 frag_ct(const bf16_t* p, int hl) {
  bf16x16 b;
#pragma unroll
  for (int e = 0; e < 16; ++e) b[e] = p[(hl << 4) + e];
  return b;
}
// B-fragment strided (K-major storage): b[e] = p[(hl*16+e)*stride]
__device__ __forceinline__ bf16x16 frag_bs(const bf16_t* p, int stride, int hl) {
  bf16x16 b;
#pragma unroll
  for (int e = 0; e < 16; ++e) b[e] = p[((hl << 4) + e) * stride];
  return b;
}

// CDNA5 async global->LDS copy (ASYNCcnt-tracked), 16B per lane
__device__ __forceinline__ void async_copy_b128(unsigned lds_byte_off, const void* gaddr) {
  asm volatile("global_load_async_to_lds_b128 %0, %1, off"
               :: "v"(lds_byte_off), "v"(gaddr) : "memory");
}
__device__ __forceinline__ void wait_async0() {
  asm volatile("s_wait_asynccnt 0x0" ::: "memory");
}

// ---------------- generic bf16 WMMA GEMM ----------------
// C[M,N] = A[M,K]*B[K,N]; fp32 inputs converted to bf16 in LDS; fp32 or bf16 out.
#define GTM 128
#define GTN 128
#define GTK 32
__launch_bounds__(256)
__global__ void gemm_kernel(const float* __restrict__ A, const float* __restrict__ B,
                            void* __restrict__ C, int M, int N, int Kd,
                            int lda, int ldb, int ldc, int obf) {
  __shared__ bf16_t sA[GTM * GTK];  // 8 KB, [m][k]
  __shared__ bf16_t sB[GTN * GTK];  // 8 KB, [n][k]  (transposed -> contiguous frags)
  const int tid = threadIdx.x;
  const int bm = blockIdx.y * GTM, bn = blockIdx.x * GTN;
  const int wid = tid >> 5, lane = tid & 31;
  const int hl = lane >> 4, lan = lane & 15;
  const int wm = (wid >> 2) * 64;
  const int wn = (wid & 3) * 32;

  f32x8 acc[4][2];
#pragma unroll
  for (int i = 0; i < 4; ++i)
#pragma unroll
    for (int j = 0; j < 2; ++j) acc[i][j] = f32x8_zero();

  for (int k0 = 0; k0 < Kd; k0 += GTK) {
    if (k0 + GTK < Kd) {  // global_prefetch_b8 next tiles
      if (tid < 128 && bm + tid < M)
        __builtin_prefetch(&A[(long)(bm + tid) * lda + k0 + GTK], 0, 1);
      if (tid < 32)
        __builtin_prefetch(&B[(long)(k0 + GTK + tid) * ldb + bn], 0, 1);
    }
    for (int i = tid; i < GTM * GTK; i += 256) {
      int r = i >> 5, c = i & 31;
      int gr = bm + r, gc = k0 + c;
      float v = (gr < M && gc < Kd) ? A[(long)gr * lda + gc] : 0.f;
      sA[r * GTK + c] = (bf16_t)v;
    }
    for (int i = tid; i < GTK * GTN; i += 256) {
      int r = i >> 7, c = i & 127;  // r = k, c = n
      int gr = k0 + r, gc = bn + c;
      float v = (gr < Kd && gc < N) ? B[(long)gr * ldb + gc] : 0.f;
      sB[c * GTK + r] = (bf16_t)v;
    }
    __syncthreads();
    bf16x16 af[4], bfg[2];
#pragma unroll
    for (int mt = 0; mt < 4; ++mt)
      af[mt] = frag_a(&sA[(wm + mt * 16 + lan) * GTK], hl);
#pragma unroll
    for (int nt = 0; nt < 2; ++nt)
      bfg[nt] = frag_ct(&sB[(wn + nt * 16 + lan) * GTK], hl);
#pragma unroll
    for (int mt = 0; mt < 4; ++mt)
#pragma unroll
      for (int nt = 0; nt < 2; ++nt) acc[mt][nt] = wmma_bf16(af[mt], bfg[nt], acc[mt][nt]);
    __syncthreads();
  }
#pragma unroll
  for (int mt = 0; mt < 4; ++mt)
#pragma unroll
    for (int nt = 0; nt < 2; ++nt)
#pragma unroll
      for (int r = 0; r < 8; ++r) {
        int row = bm + wm + mt * 16 + hl * 8 + r;
        int col = bn + wn + nt * 16 + lan;
        if (row < M && col < N) {
          if (obf) ((bf16_t*)C)[(long)row * ldc + col] = (bf16_t)acc[mt][nt][r];
          else     ((float*)C)[(long)row * ldc + col] = acc[mt][nt][r];
        }
      }
}

// ---------------- RMSNorm ----------------
__launch_bounds__(256)
__global__ void rmsnorm_kernel(const float* __restrict__ x, const float* __restrict__ w,
                               float* __restrict__ out) {
  __shared__ float red[256];
  const long s = blockIdx.x;
  float acc = 0.f;
  for (int d = threadIdx.x; d < D_; d += 256) {
    float v = x[s * D_ + d];
    acc += v * v;
  }
  red[threadIdx.x] = acc;
  __syncthreads();
  for (int st = 128; st > 0; st >>= 1) {
    if (threadIdx.x < st) red[threadIdx.x] += red[threadIdx.x + st];
    __syncthreads();
  }
  float inv = rsqrtf(red[0] / (float)D_ + EPS_);
  for (int d = threadIdx.x; d < D_; d += 256) out[s * D_ + d] = w[d] * x[s * D_ + d] * inv;
}

// ---------------- block means (bf16 q/k) ----------------
__launch_bounds__(128)
__global__ void means_kernel(const bf16_t* __restrict__ q, const bf16_t* __restrict__ k,
                             float* __restrict__ qm, float* __restrict__ km) {
  const int n = blockIdx.x, h = blockIdx.y, d = threadIdx.x;
  float sq = 0.f, sk = 0.f;
  for (int r = 0; r < BS_; ++r) {
    long idx = (long)(n * BS_ + r) * (H_ * HD_) + h * HD_ + d;
    sq += (float)q[idx];
    sk += (float)k[idx];
  }
  qm[(h * NB_ + n) * HD_ + d] = sq * (1.f / BS_);
  km[(h * NB_ + n) * HD_ + d] = sk * (1.f / BS_);
}

// ---------------- top-2 block selection ----------------
__launch_bounds__(64)
__global__ void topk_kernel(const float* __restrict__ qm, const float* __restrict__ km,
                            int* __restrict__ sel_idx, int* __restrict__ sel_valid) {
  int t = threadIdx.x;
  if (t >= H_ * NB_) return;
  int h = t >> 2, i = t & 3;
  float sc[NB_];
  for (int j = 0; j < NB_; ++j) {
    float d = 0.f;
    for (int e = 0; e < HD_; ++e)
      d += qm[(h * NB_ + i) * HD_ + e] * km[(h * NB_ + j) * HD_ + e];
    bool allowed = (j < i);
    sc[j] = allowed ? d * SCALE_ : NEG_;
  }
  int i0 = 0;
  for (int j = 1; j < NB_; ++j)
    if (sc[j] > sc[i0]) i0 = j;
  int i1 = -1;
  for (int j = 0; j < NB_; ++j) {
    if (j == i0) continue;
    if (i1 < 0 || sc[j] > sc[i1]) i1 = j;
  }
  sel_idx[t * KSEL + 0] = i0;
  sel_idx[t * KSEL + 1] = i1;
  sel_valid[t * KSEL + 0] = (i0 < i) ? 1 : 0;
  sel_valid[t * KSEL + 1] = (i1 < i) ? 1 : 0;
}

// ---------------- attention ----------------
// grid (qt=16, n=4, h=16), 256 threads / 8 waves.
// Q fragments live in registers; score B-fragments stream from global bf16;
// scores+softmax in 200KB LDS; PV V-tiles staged via async global->LDS copies.
__launch_bounds__(256)
__global__ void attn_kernel(const bf16_t* __restrict__ q16, const bf16_t* __restrict__ k16,
                            const bf16_t* __restrict__ v16, const bf16_t* __restrict__ kl16,
                            const bf16_t* __restrict__ vl16,
                            const int* __restrict__ sel_idx, const int* __restrict__ sel_valid,
                            float* __restrict__ o) {
  extern __shared__ char smem[];
  float*  sS = (float*)smem;                             // QT_*CT_ fp32 (200 KB)
  bf16_t* sV = (bf16_t*)(smem + (size_t)QT_ * CT_ * 4);  // 32*128 bf16 (8 KB)
  float*  red  = (float*)(smem + (size_t)QT_ * CT_ * 4 + 32 * 128 * 2);
  float*  rowv = red + QT_ * 8;
  const unsigned sV_off = (unsigned)QT_ * CT_ * 4;       // LDS byte offset of sV

  const int tid = threadIdx.x;
  const int qt = blockIdx.x, n = blockIdx.y, h = blockIdx.z;
  const int wid = tid >> 5, lane = tid & 31, hl = lane >> 4, lan = lane & 15;
  const int q0 = n * BS_ + qt * QT_;
  const int ld = H_ * HD_;
  const int hs = h * HD_;

  int idxS[KSEL], valS[KSEL];
#pragma unroll
  for (int s = 0; s < KSEL; ++s) {
    idxS[s] = sel_idx[(h * NB_ + n) * KSEL + s];
    valS[s] = sel_valid[(h * NB_ + n) * KSEL + s];
  }

  // Q A-fragments (2 mtiles x 4 k-steps) loaded once, reused for all 13 chunks
  bf16x16 aq[2][4];
#pragma unroll
  for (int mt = 0; mt < 2; ++mt)
#pragma unroll
    for (int kd = 0; kd < 4; ++kd)
      aq[mt][kd] = frag_a(q16 + (long)(q0 + mt * 16 + lan) * ld + hs + kd * 32, hl);

  // ---- score phase: 12 chunks of 128 keys, no LDS staging, no barriers ----
  for (int c = 0; c < 12; ++c) {
    int seg = c >> 2, cc = c & 3;
    int colbase = seg * 512 + cc * 128;
    bool valid = true;
    int kb = 0;
    if (seg == 0) {
      kb = n * BS_ + cc * 128;
    } else {
      int s = seg - 1;
      valid = (valS[s] != 0);
      kb = idxS[s] * BS_ + cc * 128;
    }
    if (!valid) {  // block-uniform
      for (int i = tid; i < QT_ * 128; i += 256) {
        int r = i >> 7, col = i & 127;
        sS[r * CT_ + colbase + col] = NEG_;
      }
      continue;
    }
    for (int tt = 0; tt < 2; ++tt) {
      int t = wid + tt * 8;
      int mt = t & 1, ntile = t >> 1;
      const bf16_t* brow = k16 + (long)(kb + ntile * 16 + lan) * ld + hs;
      f32x8 accs = f32x8_zero();
#pragma unroll
      for (int kd = 0; kd < 4; ++kd)
        accs = wmma_bf16(aq[mt][kd], frag_ct(brow + kd * 32, hl), accs);
#pragma unroll
      for (int r = 0; r < 8; ++r) {
        int row = mt * 16 + hl * 8 + r;
        int col = ntile * 16 + lan;
        float val = accs[r] * SCALE_;
        if (seg == 0) {
          int qi = qt * QT_ + row, ki = cc * 128 + col;
          if (qi < ki) val = NEG_;
        }
        sS[row * CT_ + colbase + col] = val;
      }
    }
  }
  // latent chunk: 64 cols at 1536; 8 tiles, one per wave
  {
    int t = wid;
    int mt = t & 1, ntile = t >> 1;  // ntile 0..3
    const bf16_t* brow = kl16 + (long)(ntile * 16 + lan) * ld + hs;
    f32x8 accs = f32x8_zero();
#pragma unroll
    for (int kd = 0; kd < 4; ++kd)
      accs = wmma_bf16(aq[mt][kd], frag_ct(brow + kd * 32, hl), accs);
#pragma unroll
    for (int r = 0; r < 8; ++r) {
      int row = mt * 16 + hl * 8 + r;
      int col = ntile * 16 + lan;
      sS[row * CT_ + 1536 + col] = accs[r] * SCALE_;
    }
  }
  __syncthreads();

  // ---- softmax over CT_ columns, 8 threads per row ----
  {
    int row = tid >> 3, sub = tid & 7;
    float m = -3.4e38f;
    for (int ci = sub; ci < CT_; ci += 8) m = fmaxf(m, sS[row * CT_ + ci]);
    red[row * 8 + sub] = m;
    __syncthreads();
    if (sub == 0) {
      float mm = red[row * 8];
      for (int j = 1; j < 8; ++j) mm = fmaxf(mm, red[row * 8 + j]);
      rowv[row] = mm;
    }
    __syncthreads();
    float mm = rowv[row], ssum = 0.f;
    for (int ci = sub; ci < CT_; ci += 8) {
      float e = __expf(sS[row * CT_ + ci] - mm);
      sS[row * CT_ + ci] = e;
      ssum += e;
    }
    red[row * 8 + sub] = ssum;
    __syncthreads();
    if (sub == 0) {
      float s2 = 0.f;
      for (int j = 0; j < 8; ++j) s2 += red[row * 8 + j];
      rowv[row] = 1.f / s2;
    }
    __syncthreads();
    float inv = rowv[row];
    for (int ci = sub; ci < CT_; ci += 8) sS[row * CT_ + ci] *= inv;
    __syncthreads();
  }

  // ---- PV: O(32x128) = P(32x1600) * V(1600x128), 50 chunks of 32 rows ----
  f32x8 oacc[2];
  oacc[0] = f32x8_zero();
  oacc[1] = f32x8_zero();
  for (int kc = 0; kc < 50; ++kc) {
    int cb = kc * 32;
    bool skip = false;
    const bf16_t* src = v16;
    long base = 0;
    if (cb < 512) {
      base = (long)(n * BS_ + cb);
    } else if (cb < 1536) {
      int s = (cb - 512) >> 9;
      if (!valS[s]) skip = true;  // exp(NEG)==0 -> contribution is zero
      else base = (long)(idxS[s] * BS_ + ((cb - 512) & 511));
    } else {
      src = vl16;
      base = cb - 1536;
    }
    __syncthreads();  // previous chunk's reads of sV complete
    if (!skip) {
      // 32 rows x 256B = 512 x b128 async copies; 2 per thread
      for (int i = tid; i < 512; i += 256) {
        int r = i >> 4, cby = (i & 15) * 16;
        async_copy_b128(sV_off + r * 256 + cby,
                        (const char*)(src + (base + r) * ld + hs) + cby);
      }
      wait_async0();
    }
    __syncthreads();
    if (skip) continue;
    for (int tt = 0; tt < 2; ++tt) {
      int t = wid + tt * 8;
      int mt = t & 1, ntile = t >> 1;
      bf16x16 a;
#pragma unroll
      for (int e = 0; e < 16; ++e) {
        int kk = (hl << 4) + e;
        a[e] = (bf16_t)sS[(mt * 16 + lan) * CT_ + cb + kk];
      }
      bf16x16 b = frag_bs(&sV[ntile * 16 + lan], 128, hl);
      oacc[tt] = wmma_bf16(a, b, oacc[tt]);
    }
  }
  for (int tt = 0; tt < 2; ++tt) {
    int t = wid + tt * 8;
    int mt = t & 1, ntile = t >> 1;
#pragma unroll
    for (int r = 0; r < 8; ++r) {
      int row = q0 + mt * 16 + hl * 8 + r;
      int col = ntile * 16 + lan;
      o[(long)row * ld + hs + col] = oacc[tt][r];
    }
  }
}

// ---------------- gate ----------------
__launch_bounds__(256)
__global__ void gate_kernel(const float* __restrict__ hbuf, const float* __restrict__ wg,
                            float* __restrict__ o) {
  __shared__ float red[256];
  const long s = blockIdx.x;
  float acc = 0.f;
  for (int d = threadIdx.x; d < D_; d += 256) acc += hbuf[s * D_ + d] * wg[d];
  red[threadIdx.x] = acc;
  __syncthreads();
  for (int st = 128; st > 0; st >>= 1) {
    if (threadIdx.x < st) red[threadIdx.x] += red[threadIdx.x + st];
    __syncthreads();
  }
  float g = 1.f / (1.f + __expf(-red[0]));
  float gv = (g > TAU_) ? g : 0.f;
  for (int d = threadIdx.x; d < H_ * DV_; d += 256) o[s * (H_ * DV_) + d] *= gv;
}

// ---------------- SwiGLU activation in-place on u(S, 2*DFF) ----------------
__launch_bounds__(256)
__global__ void act_kernel(float* __restrict__ u) {
  long i = (long)blockIdx.x * 256 + threadIdx.x;
  if (i >= (long)S_ * DFF_) return;
  long sr = i / DFF_, j = i % DFF_;
  float a = u[sr * (2 * DFF_) + j];
  float b = u[sr * (2 * DFF_) + DFF_ + j];
  float sb = 1.f / (1.f + __expf(-b));
  float sa = a / (1.f + __expf(-a));
  u[sr * (2 * DFF_) + j] = sb * sa;
}

// ---------------- elementwise add ----------------
__launch_bounds__(256)
__global__ void add_kernel(const float* __restrict__ a, const float* __restrict__ b,
                           float* __restrict__ c, long n) {
  long i = (long)blockIdx.x * 256 + threadIdx.x;
  if (i < n) c[i] = a[i] + b[i];
}

// ---------------- host orchestration ----------------
extern "C" void kernel_launch(void* const* d_in, const int* in_sizes, int n_in,
                              void* d_out, int out_size, void* d_ws, size_t ws_size,
                              hipStream_t stream) {
  (void)in_sizes; (void)n_in; (void)out_size; (void)ws_size;
  const float* x      = (const float*)d_in[0];
  const float* latent = (const float*)d_in[1];
  const float* n1w    = (const float*)d_in[2];
  const float* n2w    = (const float*)d_in[3];
  const float* Wq     = (const float*)d_in[4];
  const float* Wk     = (const float*)d_in[5];
  const float* Wv     = (const float*)d_in[6];
  const float* Wo     = (const float*)d_in[7];
  const float* Wlk    = (const float*)d_in[8];
  const float* Wlv    = (const float*)d_in[9];
  const float* wgate  = (const float*)d_in[10];
  const float* Wup    = (const float*)d_in[11];
  const float* Wdn    = (const float*)d_in[12];
  float* out = (float*)d_out;

  const long SD = (long)S_ * D_;
  char* w = (char*)d_ws;
  float*  hbuf = (float*)w;    w += SD * 4;
  bf16_t* q16  = (bf16_t*)w;   w += SD * 2;
  bf16_t* k16  = (bf16_t*)w;   w += SD * 2;
  bf16_t* v16  = (bf16_t*)w;   w += SD * 2;
  bf16_t* kl16 = (bf16_t*)w;   w += (long)R_ * D_ * 2;
  bf16_t* vl16 = (bf16_t*)w;   w += (long)R_ * D_ * 2;
  float*  ob   = (float*)w;    w += SD * 4;
  float*  x2   = (float*)w;    w += SD * 4;
  float*  qm   = (float*)w;    w += H_ * NB_ * HD_ * 4;
  float*  km   = (float*)w;    w += H_ * NB_ * HD_ * 4;
  int*    seli = (int*)w;      w += H_ * NB_ * KSEL * 4;
  int*    selv = (int*)w;      w += H_ * NB_ * KSEL * 4;
  float*  u    = (float*)w;    w += (long)S_ * 2 * DFF_ * 4;
  float*  down = ob;  // reuse after Wo projection consumed ob

  // 1. h = rms(x, norm1_w)
  rmsnorm_kernel<<<S_, 256, 0, stream>>>(x, n1w, hbuf);
  // 2. q/k/v = h @ W{q,k,v}  (bf16 outputs)
  gemm_kernel<<<dim3(D_ / GTN, S_ / GTM), 256, 0, stream>>>(hbuf, Wq, q16, S_, D_, D_, D_, D_, D_, 1);
  gemm_kernel<<<dim3(D_ / GTN, S_ / GTM), 256, 0, stream>>>(hbuf, Wk, k16, S_, D_, D_, D_, D_, D_, 1);
  gemm_kernel<<<dim3(D_ / GTN, S_ / GTM), 256, 0, stream>>>(hbuf, Wv, v16, S_, D_, D_, D_, D_, D_, 1);
  // 3. latent projections (bf16 outputs)
  gemm_kernel<<<dim3(D_ / GTN, 1), 256, 0, stream>>>(latent, Wlk, kl16, R_, D_, D_, D_, D_, D_, 1);
  gemm_kernel<<<dim3(D_ / GTN, 1), 256, 0, stream>>>(latent, Wlv, vl16, R_, D_, D_, D_, D_, D_, 1);
  // 4. block means + top-2 selection
  means_kernel<<<dim3(NB_, H_), 128, 0, stream>>>(q16, k16, qm, km);
  topk_kernel<<<1, 64, 0, stream>>>(qm, km, seli, selv);
  // 5. attention
  size_t attn_lds = (size_t)QT_ * CT_ * 4 + 32 * 128 * 2 + QT_ * 8 * 4 + QT_ * 4;
  attn_kernel<<<dim3(BS_ / QT_, NB_, H_), 256, attn_lds, stream>>>(
      q16, k16, v16, kl16, vl16, seli, selv, ob);
  // 6. gate
  gate_kernel<<<S_, 256, 0, stream>>>(hbuf, wgate, ob);
  // 7. x2 = x + ob @ Wo
  gemm_kernel<<<dim3(D_ / GTN, S_ / GTM), 256, 0, stream>>>(ob, Wo, x2, S_, D_, D_, D_, D_, D_, 0);
  add_kernel<<<(unsigned)((SD + 255) / 256), 256, 0, stream>>>(x, x2, x2, SD);
  // 8. h = rms(x2, norm2_w)
  rmsnorm_kernel<<<S_, 256, 0, stream>>>(x2, n2w, hbuf);
  // 9. u = h @ ffn_up
  gemm_kernel<<<dim3((2 * DFF_) / GTN, S_ / GTM), 256, 0, stream>>>(
      hbuf, Wup, u, S_, 2 * DFF_, D_, D_, 2 * DFF_, 2 * DFF_, 0);
  // 10. SwiGLU in-place
  act_kernel<<<(unsigned)(((long)S_ * DFF_ + 255) / 256), 256, 0, stream>>>(u);
  // 11. down = act @ ffn_down
  gemm_kernel<<<dim3(D_ / GTN, S_ / GTM), 256, 0, stream>>>(
      u, Wdn, down, S_, D_, DFF_, 2 * DFF_, D_, D_, 0);
  // 12. out = x2 + down
  add_kernel<<<(unsigned)((SD + 255) / 256), 256, 0, stream>>>(x2, down, out, SD);
}
